// GNNTARnetProbModel_5557687681683
// MI455X (gfx1250) — matrix operand order repaired and checked by reading, not compiled, not run on hardware
//
#include <hip/hip_runtime.h>
#include <hip/hip_bf16.h>

typedef __attribute__((ext_vector_type(16))) _Float16 v16h;
typedef __attribute__((ext_vector_type(2)))  _Float16 h2;
typedef __attribute__((ext_vector_type(2)))  float    f2;
typedef __attribute__((ext_vector_type(8)))  float    v8f;
typedef int v4i __attribute__((vector_size(16)));
typedef __attribute__((address_space(1))) v4i* gv4i_p;
typedef __attribute__((address_space(3))) v4i* sv4i_p;

#define BB 8
#define NN 5000
#define EE 160000
#define DD 64
#define HH 128
#define NK (NN * DD)   // 320000

#if __has_builtin(__builtin_amdgcn_global_load_async_to_lds_b128) && \
    __has_builtin(__builtin_amdgcn_s_wait_asynccnt)
#define USE_ASYNC_LDS 1
#endif

// ---------------------------------------------------------------------------
// WMMA 16x16x32 f16 fragment layouts (wave32, CDNA5 ISA 7.12.2), stored
// pre-swizzled in LDS so every fragment is one contiguous 32B v16h load.
//
// A element (row m, K): chunk c=K>>5; kk=K&31; hi=(kk>>3)&1; jj=(kk&7)+((kk>>4)<<3)
//   lds index = c*512 + (hi*16 + m)*16 + jj           (lane L reads [c*512 + L*16])
// B fragment f=(kchunk*4+ntile): lane L, half j holds w[kchunk*32+(L>>4)*8+(j&7)+((j>>3)<<4)][ntile*16+(L&15)]
//   lds index = (f*32 + L)*16 + j
// C/D 16x16 f32: lane L -> col N=L&15 ; elem r -> row M = r + (L>>4)*8
// ---------------------------------------------------------------------------
__device__ __forceinline__ int a_addr(int m, int K) {
  int c  = K >> 5;
  int kk = K & 31;
  int hi = (kk >> 3) & 1;
  int jj = (kk & 7) + ((kk >> 4) << 3);
  return c * 512 + (hi * 16 + m) * 16 + jj;
}

__device__ __forceinline__ void stage_w_sw(_Float16* dst, const float* __restrict__ w,
                                           int K, int tid) {
  const int total = K * 64;   // = nfrags*512
  for (int i = tid; i < total; i += 256) {
    int f = i >> 9, rem = i & 511;
    int l = rem >> 4, j = rem & 15;
    int kchunk = f >> 2, ntile = f & 3;
    int col  = ntile * 16 + (l & 15);
    int krow = kchunk * 32 + (l >> 4) * 8 + (j & 7) + ((j >> 3) << 4);
    dst[i] = (_Float16)w[krow * 64 + col];
  }
}

__device__ __forceinline__ v16h load_b_sw(const _Float16* wsw, int kchunk, int ntile, int lane) {
  return *(const v16h*)(wsw + (((kchunk * 4 + ntile) * 32 + lane) << 4));
}

__device__ __forceinline__ v16h load_a_sw(const _Float16* aS, int kchunk, int lane) {
  return *(const v16h*)(aS + kchunk * 512 + (lane << 4));
}

__device__ __forceinline__ float elu1(float v) {
  return v > 0.f ? v : (__expf(v) - 1.f);
}

// ---------------------------------------------------------------------------
__global__ __launch_bounds__(256) void k_scale_emb(const float* __restrict__ x,
                                                   const float* __restrict__ emb,
                                                   float* __restrict__ h) {
  int i = blockIdx.x * 256 + threadIdx.x;
  if (i >= BB * NN * DD) return;
  int d = i & (DD - 1);
  int bn = i >> 6;
  int n = bn % NN;
  h[i] = x[bn] * emb[n * DD + d];
}

__global__ __launch_bounds__(256) void k_zero(float* __restrict__ p, int n) {
  int i = blockIdx.x * 256 + threadIdx.x;
  if (i < n) p[i] = 0.f;
}

// ---------------------------------------------------------------------------
// Edge message FFN + scatter-add.  grid: (E/128, B); 8 waves x 16 edges.
// ---------------------------------------------------------------------------
__global__ __launch_bounds__(256) void k_edge_msg(const float* __restrict__ h,
                                                  const int* __restrict__ edges,
                                                  const float* __restrict__ pw1,
                                                  const float* __restrict__ pw2,
                                                  float* __restrict__ agg) {
  __shared__ alignas(32) _Float16 w1sw[DD * DD];      // 8 KB
  __shared__ alignas(32) _Float16 w2sw[DD * DD];      // 8 KB
  __shared__ alignas(32) _Float16 aS[8][2 * 512];     // 16 KB (2 k-chunks / wave)

  const int tid  = threadIdx.x;
  const int wave = tid >> 5;
  const int lane = tid & 31;
  const int b    = blockIdx.y;

  stage_w_sw(w1sw, pw1, DD, tid);
  stage_w_sw(w2sw, pw2, DD, tid);

  const int ebase = blockIdx.x * 128 + wave * 16;
  const float* hb = h + (size_t)b * NN * DD;

  // gather 16 parent rows; lane covers cols {2L, 2L+1}; store packed h2 at
  // its A-fragment address (pairs of K are always contiguous halves).
#pragma unroll 4
  for (int i = 0; i < 16; ++i) {
    int e   = min(ebase + i, EE - 1);
    int src = edges[2 * e];
    f2 v = *(const f2*)(hb + (size_t)src * DD + 2 * lane);
    h2 p; p[0] = (_Float16)v[0]; p[1] = (_Float16)v[1];
    *(h2*)&aS[wave][a_addr(i, 2 * lane)] = p;
  }
  __syncthreads();

  v16h a0 = load_a_sw(aS[wave], 0, lane);
  v16h a1 = load_a_sw(aS[wave], 1, lane);

  v8f c1[4];
#pragma unroll
  for (int t = 0; t < 4; ++t) {
    v8f acc = {};
    acc = __builtin_amdgcn_wmma_f32_16x16x32_f16(false, a0, false,
            load_b_sw(w1sw, 0, t, lane), (short)0, acc, false, false);
    acc = __builtin_amdgcn_wmma_f32_16x16x32_f16(false, a1, false,
            load_b_sw(w1sw, 1, t, lane), (short)0, acc, false, false);
#pragma unroll
    for (int r = 0; r < 8; ++r) acc[r] = elu1(acc[r]);
    c1[t] = acc;
  }
  __syncthreads();                 // all aS reads complete

  // re-fragment layer-1 output as an A matrix (K = output col)
#pragma unroll
  for (int t = 0; t < 4; ++t) {
    int col = t * 16 + (lane & 15);
#pragma unroll
    for (int r = 0; r < 8; ++r) {
      int mrow = r + (lane >> 4) * 8;
      aS[wave][a_addr(mrow, col)] = (_Float16)c1[t][r];
    }
  }
  __syncthreads();

  a0 = load_a_sw(aS[wave], 0, lane);
  a1 = load_a_sw(aS[wave], 1, lane);

  float* aggb = agg + (size_t)b * NN * DD;
#pragma unroll
  for (int t = 0; t < 4; ++t) {
    v8f acc = {};
    acc = __builtin_amdgcn_wmma_f32_16x16x32_f16(false, a0, false,
            load_b_sw(w2sw, 0, t, lane), (short)0, acc, false, false);
    acc = __builtin_amdgcn_wmma_f32_16x16x32_f16(false, a1, false,
            load_b_sw(w2sw, 1, t, lane), (short)0, acc, false, false);
    int col = t * 16 + (lane & 15);
#pragma unroll
    for (int r = 0; r < 8; ++r) {
      int mrow = r + (lane >> 4) * 8;
      int e = ebase + mrow;
      if (e < EE) {
        int dst = edges[2 * e + 1];
        atomicAdd(&aggb[(size_t)dst * DD + col], elu1(acc[r]));
      }
    }
  }
}

// ---------------------------------------------------------------------------
// Node update: hNext = elu([h|agg] @ uw1 + ub1) @ uw2 + ub2.  grid: (N/128, B)
// ---------------------------------------------------------------------------
__global__ __launch_bounds__(256) void k_node_update(const float* __restrict__ h,
                                                     const float* __restrict__ agg,
                                                     const float* __restrict__ uw1,  // [128][64]
                                                     const float* __restrict__ ub1,
                                                     const float* __restrict__ uw2,  // [64][64]
                                                     const float* __restrict__ ub2,
                                                     float* __restrict__ hNext) {
  __shared__ alignas(32) _Float16 w1sw[2 * DD * DD];  // 16 KB (K=128 -> 16 frags)
  __shared__ alignas(32) _Float16 w2sw[DD * DD];      // 8 KB
  __shared__ alignas(32) _Float16 aS[8][4 * 512];     // 32 KB (4 k-chunks / wave)

  const int tid  = threadIdx.x;
  const int wave = tid >> 5;
  const int lane = tid & 31;
  const int b    = blockIdx.y;

  stage_w_sw(w1sw, uw1, 2 * DD, tid);
  stage_w_sw(w2sw, uw2, DD, tid);

  const int nbase = blockIdx.x * 128 + wave * 16;
  const float* hb = h   + (size_t)b * NN * DD;
  const float* gb = agg + (size_t)b * NN * DD;

#pragma unroll 4
  for (int i = 0; i < 16; ++i) {
    int n = min(nbase + i, NN - 1);
    f2 v0 = *(const f2*)(hb + (size_t)n * DD + 2 * lane);
    f2 v1 = *(const f2*)(gb + (size_t)n * DD + 2 * lane);
    h2 p0; p0[0] = (_Float16)v0[0]; p0[1] = (_Float16)v0[1];
    h2 p1; p1[0] = (_Float16)v1[0]; p1[1] = (_Float16)v1[1];
    *(h2*)&aS[wave][a_addr(i, 2 * lane)]      = p0;
    *(h2*)&aS[wave][a_addr(i, DD + 2 * lane)] = p1;
  }
  __syncthreads();

  v16h a[4];
#pragma unroll
  for (int c = 0; c < 4; ++c) a[c] = load_a_sw(aS[wave], c, lane);

  v8f c1[4];
#pragma unroll
  for (int t = 0; t < 4; ++t) {
    v8f acc = {};
#pragma unroll
    for (int c = 0; c < 4; ++c)
      acc = __builtin_amdgcn_wmma_f32_16x16x32_f16(false, a[c], false,
              load_b_sw(w1sw, c, t, lane), (short)0, acc, false, false);
    float bias = ub1[t * 16 + (lane & 15)];
#pragma unroll
    for (int r = 0; r < 8; ++r) acc[r] = elu1(acc[r] + bias);
    c1[t] = acc;
  }
  __syncthreads();

#pragma unroll
  for (int t = 0; t < 4; ++t) {
    int col = t * 16 + (lane & 15);
#pragma unroll
    for (int r = 0; r < 8; ++r) {
      int mrow = r + (lane >> 4) * 8;
      aS[wave][a_addr(mrow, col)] = (_Float16)c1[t][r];
    }
  }
  __syncthreads();

  v16h a0 = load_a_sw(aS[wave], 0, lane);
  v16h a1 = load_a_sw(aS[wave], 1, lane);

  float* ob = hNext + (size_t)b * NN * DD;
#pragma unroll
  for (int t = 0; t < 4; ++t) {
    v8f acc = {};
    acc = __builtin_amdgcn_wmma_f32_16x16x32_f16(false, a0, false,
            load_b_sw(w2sw, 0, t, lane), (short)0, acc, false, false);
    acc = __builtin_amdgcn_wmma_f32_16x16x32_f16(false, a1, false,
            load_b_sw(w2sw, 1, t, lane), (short)0, acc, false, false);
    int col = t * 16 + (lane & 15);
    float bias = ub2[col];
#pragma unroll
    for (int r = 0; r < 8; ++r) {
      int mrow = r + (lane >> 4) * 8;
      int n = nbase + mrow;
      if (n < NN) ob[(size_t)n * DD + col] = acc[r] + bias;
    }
  }
}

// ---------------------------------------------------------------------------
// Head GEMV partials: acc[b,col] += sum_k phi[b,k]*wA[k,col] over one K-slice.
// Streams wA (164 MB) exactly once; phi slice staged via async copy to LDS.
// ---------------------------------------------------------------------------
#define KCH 1024
__global__ __launch_bounds__(256) void k_head_partial(const float* __restrict__ phi, // [B][NK]
                                                      const float* __restrict__ wA,  // [NK][128]
                                                      float* __restrict__ acc) {     // [B][128]
  __shared__ alignas(16) float pS[BB * KCH];          // 32 KB
  const int tid  = threadIdx.x;
  const int k0   = blockIdx.x * KCH;
  const int kend = min(NK - k0, KCH);

#ifdef USE_ASYNC_LDS
  if (kend == KCH) {
    // pure copy: async DMA global -> LDS, tracked by ASYNCcnt
    for (int i4 = tid; i4 < BB * KCH / 4; i4 += 256) {
      int bb = i4 >> 8;            // (i4*4) / KCH
      int kk = (i4 & 255) << 2;
      __builtin_amdgcn_global_load_async_to_lds_b128(
          (gv4i_p)(phi + (size_t)bb * NK + k0 + kk),
          (sv4i_p)&pS[bb * KCH + kk], 0, 0);
    }
    __builtin_amdgcn_s_wait_asynccnt(0);
  } else
#endif
  {
    for (int i = tid; i < BB * KCH; i += 256) {
      int bb = i >> 10, kk = i & (KCH - 1);
      pS[i] = (kk < kend) ? phi[(size_t)bb * NK + k0 + kk] : 0.f;
    }
  }
  __syncthreads();

  const int col  = tid & 127;
  const int half = tid >> 7;
  float a[BB] = {0.f, 0.f, 0.f, 0.f, 0.f, 0.f, 0.f, 0.f};
  const int ks = half * (KCH / 2);
  const int ke = min(ks + KCH / 2, kend);

  if (k0 + KCH < NK)   // speculative prefetch of next weight slice
    __builtin_prefetch(&wA[(size_t)(k0 + KCH + ks) * HH + col], 0, 1);

#pragma unroll 4
  for (int kk = ks; kk < ke; ++kk) {
    float w = wA[(size_t)(k0 + kk) * HH + col];
#pragma unroll
    for (int bb = 0; bb < BB; ++bb) a[bb] += pS[bb * KCH + kk] * w;
  }
#pragma unroll
  for (int bb = 0; bb < BB; ++bb) atomicAdd(&acc[bb * HH + col], a[bb]);
}

// ---------------------------------------------------------------------------
__global__ __launch_bounds__(32) void k_head_final(const float* __restrict__ acc, // [2][B][128]
                                                   const float* __restrict__ b0a,
                                                   const float* __restrict__ w0b,
                                                   const float* __restrict__ b0b,
                                                   const float* __restrict__ b1a,
                                                   const float* __restrict__ w1b,
                                                   const float* __restrict__ b1b,
                                                   float* __restrict__ out) {      // [B][4]
  int tid = threadIdx.x;
  int head = tid >> 4, b = (tid >> 1) & 7, j = tid & 1;
  const float* A  = acc + head * (BB * HH) + b * HH;
  const float* bA = head ? b1a : b0a;
  const float* wB = head ? w1b : w0b;
  const float* bB = head ? b1b : b0b;
  float s = 0.f;
  for (int c = 0; c < HH; ++c) s += elu1(A[c] + bA[c]) * wB[c * 2 + j];
  out[b * 4 + head * 2 + j] = s + bB[j];
}

// ---------------------------------------------------------------------------
extern "C" void kernel_launch(void* const* d_in, const int* in_sizes, int n_in,
                              void* d_out, int out_size, void* d_ws, size_t ws_size,
                              hipStream_t stream) {
  const float* x      = (const float*)d_in[0];
  const int*   edges  = (const int*)d_in[1];
  const float* emb    = (const float*)d_in[2];
  const float* c1_pw1 = (const float*)d_in[3];
  const float* c1_pw2 = (const float*)d_in[4];
  const float* c1_uw1 = (const float*)d_in[5];
  const float* c1_ub1 = (const float*)d_in[6];
  const float* c1_uw2 = (const float*)d_in[7];
  const float* c1_ub2 = (const float*)d_in[8];
  const float* c2_pw1 = (const float*)d_in[9];
  const float* c2_pw2 = (const float*)d_in[10];
  const float* c2_uw1 = (const float*)d_in[11];
  const float* c2_ub1 = (const float*)d_in[12];
  const float* c2_uw2 = (const float*)d_in[13];
  const float* c2_ub2 = (const float*)d_in[14];
  const float* w0a = (const float*)d_in[15];
  const float* b0a = (const float*)d_in[16];
  const float* w0b = (const float*)d_in[17];
  const float* b0b = (const float*)d_in[18];
  const float* w1a = (const float*)d_in[19];
  const float* b1a = (const float*)d_in[20];
  const float* w1b = (const float*)d_in[21];
  const float* b1b = (const float*)d_in[22];
  float* out = (float*)d_out;

  const size_t SZ = (size_t)BB * NN * DD;
  float* hA  = (float*)d_ws;
  float* agg = hA + SZ;
  float* hB  = agg + SZ;
  float* acc = hB + SZ;

  const int zeroBlocks = (int)((SZ + 255) / 256);
  dim3 edgeGrid((EE + 127) / 128, BB);
  dim3 nodeGrid((NN + 127) / 128, BB);
  const int headBlocks = (NK + KCH - 1) / KCH;

  k_scale_emb<<<zeroBlocks, 256, 0, stream>>>(x, emb, hA);

  k_zero<<<zeroBlocks, 256, 0, stream>>>(agg, (int)SZ);
  k_edge_msg<<<edgeGrid, 256, 0, stream>>>(hA, edges, c1_pw1, c1_pw2, agg);
  k_node_update<<<nodeGrid, 256, 0, stream>>>(hA, agg, c1_uw1, c1_ub1, c1_uw2, c1_ub2, hB);

  k_zero<<<zeroBlocks, 256, 0, stream>>>(agg, (int)SZ);
  k_edge_msg<<<edgeGrid, 256, 0, stream>>>(hB, edges, c2_pw1, c2_pw2, agg);
  k_node_update<<<nodeGrid, 256, 0, stream>>>(hB, agg, c2_uw1, c2_ub1, c2_uw2, c2_ub2, hA);

  k_zero<<<(2 * BB * HH + 255) / 256, 256, 0, stream>>>(acc, 2 * BB * HH);
  k_head_partial<<<headBlocks, 256, 0, stream>>>(hA, w0a, acc);
  k_head_partial<<<headBlocks, 256, 0, stream>>>(hA, w1a, acc + BB * HH);
  k_head_final<<<1, 32, 0, stream>>>(acc, b0a, w0b, b0b, b1a, w1b, b1b, out);
}
